// LocalGlobalEnvEncoder_2602750181777
// MI455X (gfx1250) — matrix operand order repaired and checked by reading, not compiled, not run on hardware
//
#include <hip/hip_runtime.h>
#include <cstdint>

#define N_NODES 8192
#define N_EDGES 262144
#define CH 256
#define OUTC 256
#define NH 4
#define DHEAD 64

typedef __attribute__((ext_vector_type(16))) __bf16 v16bf;
typedef __attribute__((ext_vector_type(8)))  float  v8f;

union FragAB { v16bf v; uint32_t u[8]; };
union FragC  { v8f   v; float    f[8]; };

// WMMA 16x16x32 bf16 register layouts (CDNA5 ISA §7.12.2)
__device__ __forceinline__ int a_koff(int j, int half) {
    return (j < 4) ? (8 * half + 2 * j) : (16 + 8 * half + 2 * (j - 4));
}
__device__ __forceinline__ int b_koff(int j, int half) {
    return (j < 4) ? (8 * half + 2 * j) : (16 + 8 * half + 2 * (j - 4));
}

// ---------------------------------------------------------------- utilities
__global__ void k_zero(float* __restrict__ p, int n) {
    int i = blockIdx.x * blockDim.x + threadIdx.x;
    if (i < n) p[i] = 0.0f;
}

__global__ void k_cvt_bf16(const float* __restrict__ s, __bf16* __restrict__ d, int n) {
    int i = blockIdx.x * blockDim.x + threadIdx.x;
    if (i < n) d[i] = (__bf16)s[i];
}

// weight_local is used as  hi @ W  -> pre-transpose so GEMM is uniformly A @ W^T
__global__ void k_transpose_cvt(const float* __restrict__ w, __bf16* __restrict__ wt) {
    int i = blockIdx.x * blockDim.x + threadIdx.x;
    if (i < CH * CH) {
        int o = i / CH, c = i % CH;
        wt[i] = (__bf16)w[c * CH + o];
    }
}

// ---------------------------------------------------------------- GCN conv
__global__ void k_degree(const int* __restrict__ adj, float* __restrict__ deg) {
    int e = blockIdx.x * blockDim.x + threadIdx.x;
    if (e < N_EDGES) unsafeAtomicAdd(&deg[adj[N_EDGES + e]], 1.0f);
}

__global__ void k_gcn_scatter(const int* __restrict__ adj, const float* __restrict__ x,
                              const float* __restrict__ deg, float* __restrict__ hi) {
    int e = blockIdx.x;
    int c = threadIdx.x;
    int r  = adj[e];
    int cl = adj[N_EDGES + e];
    float dd  = deg[cl] * deg[r];
    float val = (dd > 0.0f) ? rsqrtf(dd) : 0.0f;
    unsafeAtomicAdd(&hi[(size_t)cl * CH + c], x[(size_t)r * CH + c] * val);
}

// ---------------------------------------------------------------- WMMA GEMM
// out[M, Nout] = A[M, 256] @ W[Nout, 256]^T (+ bias).  qscale applied to
// columns n < CH (used to pre-scale q by 1/sqrt(DH) in the QKV projection).
template <bool BF16OUT>
__global__ void __launch_bounds__(256, 2)
k_gemm(const __bf16* __restrict__ A, const __bf16* __restrict__ W,
       const float* __restrict__ bias, void* __restrict__ outp,
       int Nout, float qscale) {
    __shared__ __align__(16) __bf16 lw[64 * CH];
    const int tid = threadIdx.x;
    const int n0  = blockIdx.y * 64;

    const uint4* src = (const uint4*)(W + (size_t)n0 * CH);
    uint4*       dst = (uint4*)lw;
    for (int i = tid; i < (64 * CH) / 8; i += 256) dst[i] = src[i];
    __syncthreads();

    const int wv = tid >> 5, lane = tid & 31, half = lane >> 4, lm = lane & 15;
    const int m0 = blockIdx.x * 128 + wv * 16;

    FragC acc[4];
    #pragma unroll
    for (int t = 0; t < 4; ++t)
        #pragma unroll
        for (int r = 0; r < 8; ++r) acc[t].f[r] = 0.0f;

    const __bf16* arow = A + (size_t)(m0 + lm) * CH;
    for (int kc = 0; kc < 8; ++kc) {
        const int k0 = kc * 32;
        FragAB a;
        #pragma unroll
        for (int j = 0; j < 8; ++j)
            a.u[j] = *(const uint32_t*)(arow + k0 + a_koff(j, half));
        #pragma unroll
        for (int nt = 0; nt < 4; ++nt) {
            FragAB b;
            #pragma unroll
            for (int j = 0; j < 8; ++j)
                b.u[j] = *(const uint32_t*)(lw + (nt * 16 + lm) * CH + k0 + b_koff(j, half));
            acc[nt].v = __builtin_amdgcn_wmma_f32_16x16x32_bf16(
                false, a.v, false, b.v, (short)0, acc[nt].v, false, false);
        }
    }

    #pragma unroll
    for (int nt = 0; nt < 4; ++nt) {
        const int n  = n0 + nt * 16 + lm;
        const float bv = bias ? bias[n] : 0.0f;
        const float sc = (n < CH) ? qscale : 1.0f;
        #pragma unroll
        for (int r = 0; r < 8; ++r) {
            const int m = m0 + r + 8 * half;
            const float val = (acc[nt].f[r] + bv) * sc;
            if (BF16OUT) ((__bf16*)outp)[(size_t)m * Nout + n] = (__bf16)val;
            else         ((float*)outp)[(size_t)m * Nout + n]  = val;
        }
    }
}

// ---------------------------------------------------------------- flash attention
// qkv [N, 768] bf16 (q|k|v, head h = cols h*64..); q pre-scaled by 1/8.
// One wave owns a 16-row query tile; 8 waves (block) share a 64-key K/V tile.
// Double-buffered tiles: K via CDNA5 async-copy (ASYNCcnt, issued one full
// compute phase ahead), V via register transpose into LDS. Row sums are
// accumulated with a P x ones(32x16) WMMA.  Output written directly as bf16.
// __launch_bounds__(256,2): 8 waves on one WGP = 2 waves/SIMD, so let the
// compiler use up to half the RF per wave (kills the scratch spills).
__global__ void __launch_bounds__(256, 2)
k_flash(const __bf16* __restrict__ qkv, __bf16* __restrict__ attn) {
    __shared__ __align__(16) __bf16 Kt[2][64][DHEAD];   // [key][d]       16 KB
    __shared__ __align__(16) __bf16 Vt[2][DHEAD][64];   // [d][key] (T)   16 KB
    __shared__ __align__(16) __bf16 Pst[8][16][64];     // per-wave P     16 KB

    const int tid  = threadIdx.x;
    const int h    = blockIdx.x >> 6;
    const int qblk = blockIdx.x & 63;
    const int wv = tid >> 5, lane = tid & 31, half = lane >> 4, lm = lane & 15;
    const int q0 = qblk * 128 + wv * 16;
    const int ld = 3 * CH;
    const int NT = N_NODES / 64;

    // Q fragments: 16 x 64 = two A frags (d 0..31, 32..63)
    FragAB qa[2];
    const __bf16* qrow = qkv + (size_t)(q0 + lm) * ld + h * DHEAD;
    #pragma unroll
    for (int c = 0; c < 2; ++c)
        #pragma unroll
        for (int j = 0; j < 8; ++j)
            qa[c].u[j] = *(const uint32_t*)(qrow + c * 32 + a_koff(j, half));

    // all-ones B fragment: l accumulation via WMMA (row sums in every lane)
    FragAB ones;
    #pragma unroll
    for (int j = 0; j < 8; ++j) ones.u[j] = 0x3F803F80u;

    FragC o[4], lacc;
    #pragma unroll
    for (int t = 0; t < 4; ++t)
        #pragma unroll
        for (int r = 0; r < 8; ++r) o[t].f[r] = 0.0f;
    #pragma unroll
    for (int r = 0; r < 8; ++r) lacc.f[r] = 0.0f;
    float mst[8];
    #pragma unroll
    for (int r = 0; r < 8; ++r) mst[r] = -3.0e38f;

    const int key2 = tid >> 2;           // 0..63
    const int dc   = (tid & 3) * 16;     // 0,16,32,48
    const __bf16* kbase = qkv + CH + h * DHEAD;
    const __bf16* vbase = qkv + 2 * CH + h * DHEAD;

    // prologue: start tile 0
    {
        uint32_t ldsk = (uint32_t)(uintptr_t)&Kt[0][key2][dc];
        const __bf16* gk = kbase + (size_t)key2 * ld + dc;
        asm volatile("global_load_async_to_lds_b128 %0, %1, off"
                     :: "v"(ldsk), "v"(gk) : "memory");
        asm volatile("global_load_async_to_lds_b128 %0, %1, off offset:16"
                     :: "v"(ldsk), "v"(gk) : "memory");
        const __bf16* gv = vbase + (size_t)key2 * ld + dc;
        uint4 v0 = *(const uint4*)(gv);
        uint4 v1 = *(const uint4*)(gv + 8);
        const __bf16* p0 = (const __bf16*)&v0;
        const __bf16* p1 = (const __bf16*)&v1;
        #pragma unroll
        for (int i = 0; i < 8; ++i) {
            Vt[0][dc + i][key2]     = p0[i];
            Vt[0][dc + 8 + i][key2] = p1[i];
        }
    }

    for (int it = 0; it < NT; ++it) {
        const int cur = it & 1, nxt = cur ^ 1;
        // my async K loads for tile `it` done (issued one phase ago)
        asm volatile("s_wait_asynccnt 0x0" ::: "memory");
        __syncthreads();   // tile `it` visible everywhere; buffers[nxt] free

        const bool havenext = (it + 1) < NT;
        uint4 nv0 = {}, nv1 = {};
        if (havenext) {
            const int kr = (it + 1) * 64 + key2;
            uint32_t ldsk = (uint32_t)(uintptr_t)&Kt[nxt][key2][dc];
            const __bf16* gk = kbase + (size_t)kr * ld + dc;
            asm volatile("global_load_async_to_lds_b128 %0, %1, off"
                         :: "v"(ldsk), "v"(gk) : "memory");
            asm volatile("global_load_async_to_lds_b128 %0, %1, off offset:16"
                         :: "v"(ldsk), "v"(gk) : "memory");
            const __bf16* gv = vbase + (size_t)kr * ld + dc;
            nv0 = *(const uint4*)(gv);
            nv1 = *(const uint4*)(gv + 8);
        }

        // S[16 x 64] = Q K^T (q pre-scaled)
        FragC s[4];
        #pragma unroll
        for (int nt = 0; nt < 4; ++nt) {
            #pragma unroll
            for (int r = 0; r < 8; ++r) s[nt].f[r] = 0.0f;
            #pragma unroll
            for (int c = 0; c < 2; ++c) {
                FragAB b;
                #pragma unroll
                for (int j = 0; j < 8; ++j)
                    b.u[j] = *(const uint32_t*)(&Kt[cur][nt * 16 + lm][c * 32 + b_koff(j, half)]);
                s[nt].v = __builtin_amdgcn_wmma_f32_16x16x32_bf16(
                    false, qa[c].v, false, b.v, (short)0, s[nt].v, false, false);
            }
        }

        // online softmax: one cross-lane max reduction per 64 keys
        #pragma unroll
        for (int r = 0; r < 8; ++r) {
            float mx = fmaxf(fmaxf(s[0].f[r], s[1].f[r]), fmaxf(s[2].f[r], s[3].f[r]));
            #pragma unroll
            for (int off = 1; off < 16; off <<= 1) mx = fmaxf(mx, __shfl_xor(mx, off, 32));
            const float mnew  = fmaxf(mst[r], mx);
            const float alpha = __expf(mst[r] - mnew);
            mst[r] = mnew;
            const int prow = r + 8 * half;   // D-layout row
            #pragma unroll
            for (int nt = 0; nt < 4; ++nt) {
                const float pv = __expf(s[nt].f[r] - mnew);
                Pst[wv][prow][nt * 16 + lm] = (__bf16)pv;
                o[nt].f[r] *= alpha;
            }
            lacc.f[r] *= alpha;
        }

        // P as two A fragments (keys 0..31, 32..63); O += P V ; l += P * 1
        #pragma unroll
        for (int c = 0; c < 2; ++c) {
            FragAB pa;
            #pragma unroll
            for (int j = 0; j < 8; ++j)
                pa.u[j] = *(const uint32_t*)(&Pst[wv][lm][c * 32 + a_koff(j, half)]);
            #pragma unroll
            for (int nt = 0; nt < 4; ++nt) {
                FragAB bv;
                #pragma unroll
                for (int j = 0; j < 8; ++j)
                    bv.u[j] = *(const uint32_t*)(&Vt[cur][nt * 16 + lm][c * 32 + b_koff(j, half)]);
                o[nt].v = __builtin_amdgcn_wmma_f32_16x16x32_bf16(
                    false, pa.v, false, bv.v, (short)0, o[nt].v, false, false);
            }
            lacc.v = __builtin_amdgcn_wmma_f32_16x16x32_bf16(
                false, pa.v, false, ones.v, (short)0, lacc.v, false, false);
        }

        // tail: transpose next V tile into LDS (load latency hidden by compute)
        if (havenext) {
            const __bf16* p0 = (const __bf16*)&nv0;
            const __bf16* p1 = (const __bf16*)&nv1;
            #pragma unroll
            for (int i = 0; i < 8; ++i) {
                Vt[nxt][dc + i][key2]     = p0[i];
                Vt[nxt][dc + 8 + i][key2] = p1[i];
            }
        }
    }

    #pragma unroll
    for (int r = 0; r < 8; ++r) {
        const float inv = 1.0f / lacc.f[r];   // row sum replicated in all lanes
        const int m = q0 + r + 8 * half;
        #pragma unroll
        for (int nt = 0; nt < 4; ++nt)
            attn[(size_t)m * CH + h * DHEAD + nt * 16 + lm] = (__bf16)(o[nt].f[r] * inv);
    }
}

// ---------------------------------------------------------------- LayerNorm(x + attn_proj)
__global__ void k_ln(const float* __restrict__ x, const float* __restrict__ ap,
                     const float* __restrict__ g, const float* __restrict__ bb,
                     float* __restrict__ out) {
    const int gidx = blockIdx.x * blockDim.x + threadIdx.x;
    const int row = gidx >> 5, lane = gidx & 31;
    float v[8], sum = 0.0f, sq = 0.0f;
    #pragma unroll
    for (int i = 0; i < 8; ++i) {
        const int c = lane + 32 * i;
        const float t = x[(size_t)row * CH + c] + ap[(size_t)row * CH + c];
        v[i] = t; sum += t; sq += t * t;
    }
    #pragma unroll
    for (int off = 1; off < 32; off <<= 1) {
        sum += __shfl_xor(sum, off, 32);
        sq  += __shfl_xor(sq, off, 32);
    }
    const float mean = sum * (1.0f / CH);
    const float var  = sq * (1.0f / CH) - mean * mean;
    const float rinv = rsqrtf(var + 1e-5f);
    #pragma unroll
    for (int i = 0; i < 8; ++i) {
        const int c = lane + 32 * i;
        out[(size_t)row * CH + c] = (v[i] - mean) * rinv * g[c] + bb[c];
    }
}

// ---------------------------------------------------------------- sigmoid gate combine
__global__ void k_combine(const float* __restrict__ lo, const float* __restrict__ gl,
                          const float* __restrict__ alpha, __bf16* __restrict__ out, int n) {
    const float w = 1.0f / (1.0f + __expf(-alpha[0]));
    const int i = blockIdx.x * blockDim.x + threadIdx.x;
    if (i < n) out[i] = (__bf16)(w * lo[i] + (1.0f - w) * gl[i]);
}

// ---------------------------------------------------------------- launch
extern "C" void kernel_launch(void* const* d_in, const int* in_sizes, int n_in,
                              void* d_out, int out_size, void* d_ws, size_t ws_size,
                              hipStream_t stream) {
    const float* x    = (const float*)d_in[0];
    const int*   adj  = (const int*)d_in[1];
    const float* wl   = (const float*)d_in[2];
    const float* ipw  = (const float*)d_in[3];
    const float* ipb  = (const float*)d_in[4];
    const float* opw  = (const float*)d_in[5];
    const float* opb  = (const float*)d_in[6];
    const float* lng  = (const float*)d_in[7];
    const float* lnb  = (const float*)d_in[8];
    const float* alp  = (const float*)d_in[9];
    const float* fcw  = (const float*)d_in[10];
    const float* fcb  = (const float*)d_in[11];
    float* out = (float*)d_out;

    char* p = (char*)d_ws;
    auto take = [&](size_t bytes) -> char* {
        char* r = p; p += (bytes + 255) & ~(size_t)255; return r;
    };
    const size_t NC = (size_t)N_NODES * CH;
    float*  deg    = (float*)take(N_NODES * 4);
    float*  hi     = (float*)take(NC * 4);
    __bf16* xbf    = (__bf16*)take(NC * 2);
    __bf16* hibf   = (__bf16*)take(NC * 2);
    __bf16* wltbf  = (__bf16*)take(CH * CH * 2);
    __bf16* ipwbf  = (__bf16*)take(3 * CH * CH * 2);
    __bf16* opwbf  = (__bf16*)take(CH * CH * 2);
    __bf16* fcwbf  = (__bf16*)take((size_t)OUTC * CH * 2);
    __bf16* qkvbf  = (__bf16*)take((size_t)N_NODES * 3 * CH * 2);
    __bf16* attnbf = (__bf16*)take(NC * 2);
    float*  aprojf = (float*)take(NC * 4);
    float*  localf = (float*)take(NC * 4);
    float*  globf  = (float*)take(NC * 4);
    __bf16* combbf = (__bf16*)take(NC * 2);

    // GCN conv (memory-bound, float atomics)
    k_zero<<<(N_NODES + 255) / 256, 256, 0, stream>>>(deg, N_NODES);
    k_zero<<<(int)((NC + 255) / 256), 256, 0, stream>>>(hi, (int)NC);
    k_degree<<<(N_EDGES + 255) / 256, 256, 0, stream>>>(adj, deg);
    k_gcn_scatter<<<N_EDGES, 256, 0, stream>>>(adj, x, deg, hi);

    // bf16 conversions
    k_cvt_bf16<<<(int)((NC + 255) / 256), 256, 0, stream>>>(x, xbf, (int)NC);
    k_cvt_bf16<<<(int)((NC + 255) / 256), 256, 0, stream>>>(hi, hibf, (int)NC);
    k_cvt_bf16<<<(3 * CH * CH + 255) / 256, 256, 0, stream>>>(ipw, ipwbf, 3 * CH * CH);
    k_cvt_bf16<<<(CH * CH + 255) / 256, 256, 0, stream>>>(opw, opwbf, CH * CH);
    k_cvt_bf16<<<(OUTC * CH + 255) / 256, 256, 0, stream>>>(fcw, fcwbf, OUTC * CH);
    k_transpose_cvt<<<(CH * CH + 255) / 256, 256, 0, stream>>>(wl, wltbf);

    // qkv = x @ in_proj^T + b  (bf16 out, q columns pre-scaled by 1/sqrt(DH))
    k_gemm<true><<<dim3(N_NODES / 128, 768 / 64), 256, 0, stream>>>(
        xbf, ipwbf, ipb, qkvbf, 768, 0.125f);
    // flash attention (bf16 out)
    k_flash<<<NH * (N_NODES / 128), 256, 0, stream>>>(qkvbf, attnbf);
    // out-proj, local embed
    k_gemm<false><<<dim3(N_NODES / 128, CH / 64), 256, 0, stream>>>(
        attnbf, opwbf, opb, aprojf, CH, 1.0f);
    k_gemm<false><<<dim3(N_NODES / 128, CH / 64), 256, 0, stream>>>(
        hibf, wltbf, nullptr, localf, CH, 1.0f);
    // LN(x + attn_proj), combine, final fc
    k_ln<<<N_NODES * 32 / 256, 256, 0, stream>>>(x, aprojf, lng, lnb, globf);
    k_combine<<<(int)((NC + 255) / 256), 256, 0, stream>>>(localf, globf, alp, combbf, (int)NC);
    k_gemm<false><<<dim3(N_NODES / 128, OUTC / 64), 256, 0, stream>>>(
        combbf, fcwbf, fcb, out, OUTC, 1.0f);
}
